// Attention_21208548508357
// MI455X (gfx1250) — matrix operand order repaired and checked by reading, not compiled, not run on hardware
//
#include <hip/hip_runtime.h>
#include <hip/hip_bf16.h>
#include <stdint.h>

// ---------------------------------------------------------------------------
// Problem: B=2, S=2048, E=768, H=12, d=64.
// Reference einsum 'bhij,bihd->bihd' contracts j over att only => rowsum of
// softmax == 1 => y == v.  So:   out = (x @ Wv + bv) @ Wproj + bproj
// Two 4096x768x768 GEMMs, done in bf16 WMMA with fp32 accumulation.
// ---------------------------------------------------------------------------

typedef __attribute__((ext_vector_type(16))) __bf16        v16bf;
typedef __attribute__((ext_vector_type(8)))  float         v8f;
typedef __attribute__((ext_vector_type(4)))  unsigned int  uint4v;

__device__ __forceinline__ uint16_t f2bf_rne(float f) {
    union { float f; uint32_t u; } c; c.f = f;
    uint32_t u = c.u;
    uint32_t r = u + 0x7FFFu + ((u >> 16) & 1u);   // round-to-nearest-even
    return (uint16_t)(r >> 16);
}

union FragU {
    v16bf    bf;
    uint32_t u[8];
};

// Load two 16-byte chunks (8 bf16 each) into a 16-element bf16 fragment.
__device__ __forceinline__ void load2x16B(FragU& f, const uint16_t* p, int elemOff2) {
    *reinterpret_cast<uint4v*>(&f.u[0]) = *reinterpret_cast<const uint4v*>(p);
    *reinterpret_cast<uint4v*>(&f.u[4]) = *reinterpret_cast<const uint4v*>(p + elemOff2);
}

// --------------------------- conversion kernels ----------------------------

__global__ void cvt_f32_to_bf16(const float* __restrict__ in,
                                uint16_t* __restrict__ out, int n4) {
    int idx = blockIdx.x * blockDim.x + threadIdx.x;
    if (idx < n4) {
        float4 v = reinterpret_cast<const float4*>(in)[idx];
        ushort4 o;
        o.x = f2bf_rne(v.x); o.y = f2bf_rne(v.y);
        o.z = f2bf_rne(v.z); o.w = f2bf_rne(v.w);
        reinterpret_cast<ushort4*>(out)[idx] = o;
    }
}

// WT[n*K + k] = bf16(W[k*ldw + coloff + n])   (K x N slice, transposed)
__global__ void cvt_transpose_bf16(const float* __restrict__ W,
                                   uint16_t* __restrict__ WT,
                                   int K, int ldw, int coloff, int N) {
    int idx = blockIdx.x * blockDim.x + threadIdx.x;
    if (idx < K * N) {
        int k = idx / N;
        int n = idx - k * N;
        WT[(size_t)n * K + k] = f2bf_rne(W[(size_t)k * ldw + coloff + n]);
    }
}

// ------------------------------ WMMA GEMM ----------------------------------
// C[m,n] = sum_k A[m,k] * BT[n,k] + bias[n]
// Block: 128 threads = 4 waves; block tile 64(M) x 128(N); wave tile 32x64.
// OUT_BF16=1 -> store bf16 (intermediate V), OUT_BF16=0 -> store fp32.

template <int OUT_BF16>
__global__ __launch_bounds__(128)
void gemm_bias_wmma(const uint16_t* __restrict__ A,
                    const uint16_t* __restrict__ BT,
                    const float* __restrict__ bias,
                    void* __restrict__ out,
                    int M, int N, int K) {
    const int lane = threadIdx.x & 31;
    const int wave = threadIdx.x >> 5;
    const int m0   = blockIdx.x * 64 + (wave & 1) * 32;
    const int n0   = blockIdx.y * 128 + (wave >> 1) * 64;
    const int r    = lane & 15;   // row (A) / col (B) within 16-wide tile
    const int half = lane >> 4;   // lane-half selects K sub-block

    v8f acc[2][4];
#pragma unroll
    for (int i = 0; i < 2; ++i)
#pragma unroll
        for (int j = 0; j < 4; ++j)
            acc[i][j] = (v8f)(0.0f);

    for (int k0 = 0; k0 < K; k0 += 32) {
        FragU a[2], b[4];
        // A fragment (16-bit 16x32 ISA layout): elems 0-7 @ k0+half*8,
        // elems 8-15 @ k0+16+half*8.
#pragma unroll
        for (int i = 0; i < 2; ++i) {
            const uint16_t* p = A + (size_t)(m0 + 16 * i + r) * K + (k0 + half * 8);
            load2x16B(a[i], p, 16);
        }
        // B fragment (32x16): elems 0-7 @ k0+half*8, elems 8-15 @ k0+16+half*8 of col n.
#pragma unroll
        for (int j = 0; j < 4; ++j) {
            const uint16_t* p = BT + (size_t)(n0 + 16 * j + r) * K + (k0 + half * 8);
            load2x16B(b[j], p, 16);
        }
#pragma unroll
        for (int i = 0; i < 2; ++i)
#pragma unroll
            for (int j = 0; j < 4; ++j)
                acc[i][j] = __builtin_amdgcn_wmma_f32_16x16x32_bf16(
                    false, a[i].bf, false, b[j].bf,
                    (short)0, acc[i][j], false, false);
    }

    // Epilogue: C/D layout -> lane = N (mod 16), VGPR v = M (0-7), halves at M>=8.
#pragma unroll
    for (int j = 0; j < 4; ++j) {
        const int   n  = n0 + 16 * j + r;
        const float bv = bias[n];
#pragma unroll
        for (int i = 0; i < 2; ++i) {
#pragma unroll
            for (int v = 0; v < 8; ++v) {
                const int m   = m0 + 16 * i + half * 8 + v;
                const float val = acc[i][j][v] + bv;
                if (OUT_BF16)
                    reinterpret_cast<uint16_t*>(out)[(size_t)m * N + n] = f2bf_rne(val);
                else
                    reinterpret_cast<float*>(out)[(size_t)m * N + n] = val;
            }
        }
    }
}

// ------------------------------ launcher -----------------------------------

extern "C" void kernel_launch(void* const* d_in, const int* in_sizes, int n_in,
                              void* d_out, int out_size, void* d_ws, size_t ws_size,
                              hipStream_t stream) {
    const float* x      = (const float*)d_in[0];  // [2,2048,768]
    const float* W_attn = (const float*)d_in[1];  // [768, 2304]
    const float* b_attn = (const float*)d_in[2];  // [2304]
    const float* W_proj = (const float*)d_in[3];  // [768, 768]
    const float* b_proj = (const float*)d_in[4];  // [768]

    const int M = 4096, N = 768, K = 768;

    uint16_t* xb  = (uint16_t*)d_ws;              // x in bf16          [M,K]
    uint16_t* vb  = xb  + (size_t)M * K;          // V in bf16          [M,N]
    uint16_t* wvT = vb  + (size_t)M * K;          // Wv^T in bf16       [N,K]
    uint16_t* wpT = wvT + (size_t)N * K;          // Wproj^T in bf16    [N,K]

    {
        int n4 = (M * K) / 4;
        cvt_f32_to_bf16<<<(n4 + 255) / 256, 256, 0, stream>>>(x, xb, n4);
    }
    cvt_transpose_bf16<<<(K * N + 255) / 256, 256, 0, stream>>>(
        W_attn, wvT, K, 3 * N, 2 * N, N);   // V slice: cols [1536,2304)
    cvt_transpose_bf16<<<(K * N + 255) / 256, 256, 0, stream>>>(
        W_proj, wpT, K, N, 0, N);

    dim3 grid(M / 64, N / 128);   // (64, 6)
    gemm_bias_wmma<1><<<grid, 128, 0, stream>>>(xb, wvT, b_attn + 2 * N, vb, M, N, K);
    gemm_bias_wmma<0><<<grid, 128, 0, stream>>>(vb, wpT, b_proj, d_out, M, N, K);
}
